// Dense_test_40510131536141
// MI455X (gfx1250) — matrix operand order repaired and checked by reading, not compiled, not run on hardware
//
#include <hip/hip_runtime.h>
#include <math.h>

// ---------------- problem constants ----------------
#define B_SZ   1024
#define NHID   200
#define NBR    8
#define NB     1600        // NHID * NBR
#define NOUT   12
#define TSTEPS 101
#define IN_DIM 120

typedef __attribute__((ext_vector_type(16))) __bf16 v16bf;
typedef __attribute__((ext_vector_type(8)))  __bf16 v8bf;
typedef __attribute__((ext_vector_type(8)))  float  v8f;

// ---------------- bf16 helpers (storage = ushort) ----------------
__device__ __forceinline__ unsigned short f2bf(float f) {
    union { float f; unsigned u; } v; v.f = f;
    unsigned u = v.u;
    unsigned r = u + 0x7fffu + ((u >> 16) & 1u);   // round-to-nearest-even
    if ((u & 0x7f800000u) == 0x7f800000u) r = u;   // inf/nan passthrough
    return (unsigned short)(r >> 16);
}
__device__ __forceinline__ float bf2f(unsigned short h) {
    union { unsigned u; float f; } v; v.u = ((unsigned)h) << 16;
    return v.f;
}

// ---------------- small prep kernels ----------------
__global__ void sigmoid_kernel(const float* __restrict__ in, float* __restrict__ out, int n) {
    int i = blockIdx.x * blockDim.x + threadIdx.x;
    if (i < n) out[i] = 1.0f / (1.0f + expf(-in[i]));
}

// Wt[k][n] = bf16(W[n][k]); pad rows (k >= in_f) stay zero from the initial memset.
__global__ void convert_wt_kernel(const float* __restrict__ W, unsigned short* __restrict__ Wt, int in_f) {
    int idx = blockIdx.x * blockDim.x + threadIdx.x;
    if (idx >= NB * in_f) return;
    int n = idx / in_f, k = idx % in_f;
    Wt[k * NB + n] = f2bf(W[n * in_f + k]);
}

__device__ __forceinline__ void pack_one(const float* __restrict__ x,
                                         unsigned short* __restrict__ act1, int t, int idx) {
    int b = idx / IN_DIM, j = idx % IN_DIM;
    int c = j / 40, f = j % 40;
    float v = x[((b * 3 + c) * TSTEPS + t) * 40 + f];
    act1[b * 320 + j] = f2bf(v);
}

// act1[p][b, 0:120] = bf16(x[b, c, t, f]) with j = c*40 + f
__global__ void pack_x_kernel(const float* __restrict__ x, unsigned short* __restrict__ act1, int t) {
    int idx = blockIdx.x * blockDim.x + threadIdx.x;
    if (idx >= B_SZ * IN_DIM) return;
    pack_one(x, act1, t, idx);
}

// ---------------- fused WMMA layer kernel ----------------
// Workgroup tile: 32(M=batch) x 64(N=neuron*branch); 8 waves as 2(M) x 4(N) subtiles.
// A panel [32 x KPAD] and B panel [KPAD x 64] are staged in LDS (padded pitch to
// avoid bank conflicts), then each wave runs the K loop entirely from LDS.
//   act : [B, KPAD] bf16, cols [0:RECOFF) = forward input, [RECOFF:RECOFF+200) = s(t-1)
//   Wt  : [KPAD, 1600] bf16 (transposed, zero-padded)
template<int KPAD, int NK, int RECOFF, int FWD_STRIDE>
__global__ __launch_bounds__(256)
void snn_layer_kernel(const unsigned short* __restrict__ act,
                      const unsigned short* __restrict__ Wt,
                      const float* __restrict__ bias,      // [1600]
                      const float* __restrict__ beta,      // [1600] sigmoid(tau_n)
                      const float* __restrict__ alpha,     // [200]  sigmoid(tau_m)
                      float* __restrict__ dstate,          // [B, 1600]
                      float* __restrict__ mem,             // [B, 200]
                      unsigned short* __restrict__ spike_fwd,   // [B, FWD_STRIDE] (col 0..199)
                      unsigned short* __restrict__ rec_out)     // other-parity act buffer
{
    constexpr int APITCH = KPAD + 8;   // pad keeps 16B row alignment, breaks bank conflicts
    constexpr int BPITCH = 64 + 8;
    __shared__ __align__(16) unsigned short sA[32 * APITCH];
    __shared__ __align__(16) unsigned short sB[KPAD * BPITCH];

    const int tid = threadIdx.x;
    const int tile_m = blockIdx.x & 31;     // 32 M tiles of 32 rows
    const int tile_n = blockIdx.x >> 5;     // 25 N tiles of 64 cols
    const int m0 = tile_m * 32;
    const int n0 = tile_n * 64;

    // ---- cooperative panel staging (16B chunks) ----
    {
        constexpr int CA = 32 * (KPAD / 8);
        for (int i = tid; i < CA; i += 256) {
            int row = i / (KPAD / 8);
            int col = (i % (KPAD / 8)) * 8;
            *(v8bf*)&sA[row * APITCH + col] = *(const v8bf*)&act[(m0 + row) * KPAD + col];
        }
        constexpr int CB = KPAD * 8;
        for (int i = tid; i < CB; i += 256) {
            int row = i >> 3;
            int col = (i & 7) * 8;
            *(v8bf*)&sB[row * BPITCH + col] = *(const v8bf*)&Wt[row * NB + n0 + col];
        }
    }
    __syncthreads();

    const int wid  = tid >> 5;          // 0..7
    const int lane = tid & 31;
    const int wm = wid >> 2;            // 0..1 : M subtile
    const int wn = wid & 3;             // 0..3 : N subtile

    // A fragment (16-bit A 16x32 layout):
    //  lanes 0-15 : M=lane,    K = {0..7} then {16..23}
    //  lanes 16-31: M=lane-16, K = {8..15} then {24..31}
    const int asel = (lane >> 4) * 8;
    const unsigned short* ap = &sA[(wm * 16 + (lane & 15)) * APITCH + asel];
    // B fragment (32x16): N = lane&15, K = same order as A (strided LDS gather)
    const unsigned short* bp = &sB[((lane >> 4) * 8) * BPITCH + wn * 16 + (lane & 15)];

    v8f c = {};
#pragma unroll
    for (int kk = 0; kk < NK; ++kk) {
        v8bf a_lo = *(const v8bf*)(ap);
        v8bf a_hi = *(const v8bf*)(ap + 16);
        v16bf A  = __builtin_shufflevector(a_lo, a_hi, 0,1,2,3,4,5,6,7,8,9,10,11,12,13,14,15);
        v16bf Bm;
#pragma unroll
        for (int s = 0; s < 16; ++s)
            Bm[s] = *(const __bf16*)(bp + ((s < 8) ? s : (s + 8)) * BPITCH);
        c = __builtin_amdgcn_wmma_f32_16x16x32_bf16(false, A, false, Bm,
                                                    (short)0, c, false, false);
        ap += 32;
        bp += 32 * BPITCH;
    }

    // C layout: VGPR r, lanes 0-15 -> M=r, lanes 16-31 -> M=8+r; N = lane&15.
    const int ncol   = n0 + wn * 16 + (lane & 15);
    const int mbase  = m0 + wm * 16 + ((lane >> 4) << 3);
    const int lbase  = wm * 16 + ((lane >> 4) << 3);       // local row in sA
    const float bv = bias[ncol];
    const float be = beta[ncol];
    const int neuron = ((n0 + wn * 16) >> 3) + ((lane >> 3) & 1);  // valid for writer lanes

#pragma unroll
    for (int r = 0; r < 8; ++r) {
        const int m = mbase + r;
        float ff = c[r] + bv;
        float dn = be * dstate[m * NB + ncol] + (1.0f - be) * ff;
        dstate[m * NB + ncol] = dn;

        // sum over the 8 branches of this neuron (lanes grouped by L&7)
        float s = dn;
        s += __shfl_xor(s, 1, 32);
        s += __shfl_xor(s, 2, 32);
        s += __shfl_xor(s, 4, 32);

        if ((lane & 7) == 0) {
            float al = alpha[neuron];
            float mo = mem[m * NHID + neuron];
            unsigned short sob = sA[(lbase + r) * APITCH + RECOFF + neuron]; // spike(t-1)
            float so = sob ? 1.0f : 0.0f;
            float mn = (mo - so) * al + (1.0f - al) * s;
            mem[m * NHID + neuron] = mn;
            unsigned short sb = (mn > 1.0f) ? (unsigned short)0x3F80u : (unsigned short)0u;
            spike_fwd[m * FWD_STRIDE + neuron] = sb;        // input to next layer (this t)
            rec_out[m * KPAD + RECOFF + neuron] = sb;       // recurrent input (t+1, other parity)
        }
    }
}

// ---------------- readout layer (+ fused input pack for t+1) ----------------
__global__ void layer4_pack_kernel(const unsigned short* __restrict__ s3,   // [B,200] bf16
                                   const float* __restrict__ W4,            // [12,200]
                                   const float* __restrict__ b4,            // [12]
                                   const float* __restrict__ alpha4,        // [12] sigmoid
                                   float* __restrict__ m4, float* __restrict__ acc,
                                   const float* __restrict__ x,
                                   unsigned short* __restrict__ act1_next, int t_next)
{
    int idx = blockIdx.x * blockDim.x + threadIdx.x;
    if (idx < B_SZ * NOUT) {
        int b = idx / NOUT, o = idx % NOUT;
        const unsigned short* srow = s3 + b * NHID;
        const float* wrow = W4 + o * NHID;
        float sum = b4[o];
        for (int k = 0; k < NHID; ++k) sum += bf2f(srow[k]) * wrow[k];
        float a = alpha4[o];
        float mn = m4[idx] * a + (1.0f - a) * sum;
        m4[idx] = mn;
        acc[idx] += mn;
    } else {
        int j = idx - B_SZ * NOUT;
        if (t_next < TSTEPS && j < B_SZ * IN_DIM)
            pack_one(x, act1_next, t_next, j);
    }
}

__global__ void finalize_kernel(const float* __restrict__ acc, float* __restrict__ out) {
    int b = blockIdx.x * blockDim.x + threadIdx.x;
    if (b >= B_SZ) return;
    float v[NOUT];
    float mx = -3.4e38f;
    for (int o = 0; o < NOUT; ++o) {
        v[o] = acc[b * NOUT + o] * (1.0f / (float)TSTEPS);
        mx = fmaxf(mx, v[o]);
    }
    float s = 0.0f;
    for (int o = 0; o < NOUT; ++o) s += expf(v[o] - mx);
    float lse = mx + logf(s);
    for (int o = 0; o < NOUT; ++o) out[b * NOUT + o] = v[o] - lse;
}

// ---------------- host launcher ----------------
extern "C" void kernel_launch(void* const* d_in, const int* in_sizes, int n_in,
                              void* d_out, int out_size, void* d_ws, size_t ws_size,
                              hipStream_t stream) {
    const float* x   = (const float*)d_in[0];
    const float* W1  = (const float*)d_in[1];
    const float* b1  = (const float*)d_in[2];
    const float* tm1 = (const float*)d_in[3];
    const float* tn1 = (const float*)d_in[4];
    const float* W2  = (const float*)d_in[5];
    const float* b2  = (const float*)d_in[6];
    const float* tm2 = (const float*)d_in[7];
    const float* tn2 = (const float*)d_in[8];
    const float* W3  = (const float*)d_in[9];
    const float* b3  = (const float*)d_in[10];
    const float* tm3 = (const float*)d_in[11];
    const float* tn3 = (const float*)d_in[12];
    const float* W4  = (const float*)d_in[13];
    const float* b4  = (const float*)d_in[14];
    const float* tm4 = (const float*)d_in[15];

    char* ws = (char*)d_ws;
    size_t off = 0;
    auto alloc = [&](size_t bytes) -> char* {
        char* p = ws + off;
        off = (off + bytes + 255) & ~(size_t)255;
        return p;
    };

    unsigned short* Wt1 = (unsigned short*)alloc((size_t)320 * NB * 2);
    unsigned short* Wt2 = (unsigned short*)alloc((size_t)416 * NB * 2);
    unsigned short* Wt3 = (unsigned short*)alloc((size_t)416 * NB * 2);
    unsigned short* act1[2], *act2[2], *act3[2];
    for (int p = 0; p < 2; ++p) act1[p] = (unsigned short*)alloc((size_t)B_SZ * 320 * 2);
    for (int p = 0; p < 2; ++p) act2[p] = (unsigned short*)alloc((size_t)B_SZ * 416 * 2);
    for (int p = 0; p < 2; ++p) act3[p] = (unsigned short*)alloc((size_t)B_SZ * 416 * 2);
    unsigned short* s3buf = (unsigned short*)alloc((size_t)B_SZ * NHID * 2);
    float* d1   = (float*)alloc((size_t)B_SZ * NB * 4);
    float* d2   = (float*)alloc((size_t)B_SZ * NB * 4);
    float* d3   = (float*)alloc((size_t)B_SZ * NB * 4);
    float* mem1 = (float*)alloc((size_t)B_SZ * NHID * 4);
    float* mem2 = (float*)alloc((size_t)B_SZ * NHID * 4);
    float* mem3 = (float*)alloc((size_t)B_SZ * NHID * 4);
    float* m4   = (float*)alloc((size_t)B_SZ * NOUT * 4);
    float* acc  = (float*)alloc((size_t)B_SZ * NOUT * 4);
    float* al1  = (float*)alloc(NHID * 4);
    float* al2  = (float*)alloc(NHID * 4);
    float* al3  = (float*)alloc(NHID * 4);
    float* be1  = (float*)alloc(NB * 4);
    float* be2  = (float*)alloc(NB * 4);
    float* be3  = (float*)alloc(NB * 4);
    float* al4  = (float*)alloc(NOUT * 4);
    size_t used = off;

    // zero all state + weight pads (deterministic per call)
    hipMemsetAsync(d_ws, 0, used, stream);

    // sigmoid tables
    sigmoid_kernel<<<(NHID + 255) / 256, 256, 0, stream>>>(tm1, al1, NHID);
    sigmoid_kernel<<<(NHID + 255) / 256, 256, 0, stream>>>(tm2, al2, NHID);
    sigmoid_kernel<<<(NHID + 255) / 256, 256, 0, stream>>>(tm3, al3, NHID);
    sigmoid_kernel<<<(NB + 255) / 256, 256, 0, stream>>>(tn1, be1, NB);
    sigmoid_kernel<<<(NB + 255) / 256, 256, 0, stream>>>(tn2, be2, NB);
    sigmoid_kernel<<<(NB + 255) / 256, 256, 0, stream>>>(tn3, be3, NB);
    sigmoid_kernel<<<(NOUT + 255) / 256, 256, 0, stream>>>(tm4, al4, NOUT);

    // transposed bf16 weights
    convert_wt_kernel<<<(NB * 320 + 255) / 256, 256, 0, stream>>>(W1, Wt1, 320);
    convert_wt_kernel<<<(NB * 400 + 255) / 256, 256, 0, stream>>>(W2, Wt2, 400);
    convert_wt_kernel<<<(NB * 400 + 255) / 256, 256, 0, stream>>>(W3, Wt3, 400);

    // initial input pack for t = 0
    pack_x_kernel<<<(B_SZ * IN_DIM + 255) / 256, 256, 0, stream>>>(x, act1[0], 0);

    const dim3 lgrid(800), lblk(256);   // 800 WGs = 32 M-tiles x 25 N-tiles
    const int l4grid = (B_SZ * NOUT + B_SZ * IN_DIM + 255) / 256;
    for (int t = 0; t < TSTEPS; ++t) {
        int p = t & 1, q = p ^ 1;
        snn_layer_kernel<320, 10, 120, 416><<<lgrid, lblk, 0, stream>>>(
            act1[p], Wt1, b1, be1, al1, d1, mem1, act2[p], act1[q]);
        snn_layer_kernel<416, 13, 200, 416><<<lgrid, lblk, 0, stream>>>(
            act2[p], Wt2, b2, be2, al2, d2, mem2, act3[p], act2[q]);
        snn_layer_kernel<416, 13, 200, 200><<<lgrid, lblk, 0, stream>>>(
            act3[p], Wt3, b3, be3, al3, d3, mem3, s3buf, act3[q]);
        // readout for t, fused with input pack for t+1 (writes other-parity act1)
        layer4_pack_kernel<<<l4grid, 256, 0, stream>>>(
            s3buf, W4, b4, al4, m4, acc, x, act1[q], t + 1);
    }
    finalize_kernel<<<(B_SZ + 255) / 256, 256, 0, stream>>>(acc, (float*)d_out);
}
